// VCTreeLSTMContext_25933012533728
// MI455X (gfx1250) — matrix operand order repaired and checked by reading, not compiled, not run on hardware
//
#include <hip/hip_runtime.h>
#include <hip/hip_bf16.h>
#include <math.h>

// ---------------- problem dims ----------------
#define BDIM 64
#define TDIM 64
#define BT   (BDIM*TDIM)      // 4096 rows
#define CDIM 151
#define EDIM 200
#define ROI  4096
#define HDIM 512
#define HH   256
// padded K dims (multiples of 32 for 16x16x32 bf16 WMMA)
#define K_OBJPRE 4448         // [x 4096 | obj_embed 200 | pos 128] = 4424 -> 4448
#define K_XDEC   4960         // [obj_pre 4448 | enc 512]
#define K_XEDGE  4608         // [x 4096 | enc 512]
#define K_E      224          // embedding 200 -> 224
#define K_EH     736          // decoder recurrent input [e 224 | h 512]
#define NG_HH    1024         // 4*HH gates
#define NG_H     2048         // 4*H gates

typedef __bf16 bf16;
typedef __attribute__((ext_vector_type(16))) __bf16 v16bf;
typedef __attribute__((ext_vector_type(8)))  __bf16 v8bf;
typedef __attribute__((ext_vector_type(8)))  float  v8f;

static __device__ __forceinline__ bf16 f2bf(float f) {
  unsigned u = __builtin_bit_cast(unsigned, f);
  u += 0x7fffu + ((u >> 16) & 1u);                // round-to-nearest-even
  unsigned short s = (unsigned short)(u >> 16);
  return __builtin_bit_cast(bf16, s);
}
static __device__ __forceinline__ float sigm(float x) { return 1.0f / (1.0f + __expf(-x)); }

// ---------------- WMMA GEMM: C[M,N] = A[M,K](bf16) * Wt[N,K](bf16)^T (+bias)(+C) ----------------
// REQUIRES: M % 64 == 0, N % 128 == 0, K % 32 == 0 (all call sites comply).
// Block = 256 threads = 8 waves; block tile 64x128; wave tile 16x64 (4 wmma tiles).
// No divergent guards anywhere: EXEC stays all-ones (WMMA requirement), straight-line K loop.
// Fragment layouts (wave32); this code uses contiguous k order k = 16*(l>>4) + s in both A and B:
//  A: lane l holds row m=l&15; K halves split by l>>4; two contiguous 16B loads.
//  B: lane l holds col n=l&15; K=0..15 (lanes 0-15) / 16..31 (lanes 16-31); one 32B load.
template<bool OUT_BF16, bool ACCUM>
__global__ __launch_bounds__(256)
void k_gemm(const bf16* __restrict__ A, int lda,
            const bf16* __restrict__ Wt, int ldw,
            const float* __restrict__ bias,
            void* __restrict__ Cout, int ldc,
            int K)
{
  const int wave = threadIdx.x >> 5;
  const int lane = threadIdx.x & 31;
  const int row0 = blockIdx.y * 64 + (wave & 3) * 16;
  const int col0 = blockIdx.x * 128 + (wave >> 2) * 64;
  const int lm = lane & 15, lh = lane >> 4;

  v8f acc0 = v8f{}, acc1 = v8f{}, acc2 = v8f{}, acc3 = v8f{};

  const bf16* aRow = A + (size_t)(row0 + lm) * lda + lh * 16;
  const bf16* b0 = Wt + (size_t)(col0 + lm) * ldw + lh * 16;
  const bf16* b1 = b0 + (size_t)16 * ldw;
  const bf16* b2 = b0 + (size_t)32 * ldw;
  const bf16* b3 = b0 + (size_t)48 * ldw;

  for (int k0 = 0; k0 < K; k0 += 32) {
    union { v16bf v; v8bf h[2]; } a;
    a.h[0] = *(const v8bf*)(aRow + k0);
    a.h[1] = *(const v8bf*)(aRow + k0 + 8);
    const v16bf vb0 = *(const v16bf*)(b0 + k0);
    const v16bf vb1 = *(const v16bf*)(b1 + k0);
    const v16bf vb2 = *(const v16bf*)(b2 + k0);
    const v16bf vb3 = *(const v16bf*)(b3 + k0);
    __builtin_prefetch(aRow + k0 + 128, 0, 0);   // global_prefetch_b8
    __builtin_prefetch(b0 + k0 + 128, 0, 0);
    acc0 = __builtin_amdgcn_wmma_f32_16x16x32_bf16(false, a.v, false, vb0, (short)0, acc0, false, false);
    acc1 = __builtin_amdgcn_wmma_f32_16x16x32_bf16(false, a.v, false, vb1, (short)0, acc1, false, false);
    acc2 = __builtin_amdgcn_wmma_f32_16x16x32_bf16(false, a.v, false, vb2, (short)0, acc2, false, false);
    acc3 = __builtin_amdgcn_wmma_f32_16x16x32_bf16(false, a.v, false, vb3, (short)0, acc3, false, false);
  }

#define GEMM_EPI(ACCV, J)                                                        \
  {                                                                              \
    const int n = col0 + (J) * 16 + lm;                                          \
    const float bb = bias ? bias[n] : 0.0f;                                      \
    _Pragma("unroll")                                                            \
    for (int v = 0; v < 8; ++v) {                                                \
      const int m = row0 + v + lh * 8;                                           \
      const size_t idx = (size_t)m * ldc + n;                                    \
      float val = (ACCV)[v] + bb;                                                \
      if (ACCUM) val += ((const float*)Cout)[idx];                               \
      if (OUT_BF16) ((bf16*)Cout)[idx] = f2bf(val);                              \
      else          ((float*)Cout)[idx] = val;                                   \
    }                                                                            \
  }
  GEMM_EPI(acc0, 0)
  GEMM_EPI(acc1, 1)
  GEMM_EPI(acc2, 2)
  GEMM_EPI(acc3, 3)
#undef GEMM_EPI
}

// ---------------- weight convert+transpose: dst[n, off+k] = bf16(src[row0+k, n]) (zero tail) ----------------
__global__ void k_convtr(const float* __restrict__ src, bf16* __restrict__ dst,
                         int dstLd, int dstColOff, int Kcount, int Ktot,
                         int N, int srcLd, int srcRow0)
{
  long idx = (long)blockIdx.x * blockDim.x + threadIdx.x;
  long total = (long)N * Ktot;
  if (idx >= total) return;
  int n = (int)(idx / Ktot), k = (int)(idx - (long)n * Ktot);
  float v = (k < Kcount) ? src[(size_t)(srcRow0 + k) * srcLd + n] : 0.0f;
  dst[(size_t)n * dstLd + dstColOff + k] = f2bf(v);
}

// ---------------- x (f32) -> bf16 into xdec[:,0:4096] and xedge[:,0:4096] ----------------
__global__ void k_copy_x(const float* __restrict__ x, bf16* __restrict__ xdec,
                         bf16* __restrict__ xedge)
{
  long idx = (long)blockIdx.x * 256 + threadIdx.x;
  if (idx >= (long)BT * ROI) return;
  int row = (int)(idx >> 12), c = (int)(idx & (ROI - 1));
  bf16 v = f2bf(x[idx]);
  xdec [(size_t)row * K_XDEC  + c] = v;
  xedge[(size_t)row * K_XEDGE + c] = v;
}

// ---------------- softmax(dists) @ emb1 -> obj_embed (bf16, into xdec + padded buffer) ----------------
__global__ __launch_bounds__(256)
void k_softmax_embed(const float* __restrict__ dists, const float* __restrict__ emb1,
                     bf16* __restrict__ xdec, bf16* __restrict__ objE)
{
  __shared__ float p[CDIM];
  __shared__ float stat;
  const int row = blockIdx.x, tid = threadIdx.x;
  if (tid < CDIM) p[tid] = dists[(size_t)row * CDIM + tid];
  __syncthreads();
  if (tid == 0) { float m = p[0]; for (int i = 1; i < CDIM; ++i) m = fmaxf(m, p[i]); stat = m; }
  __syncthreads();
  const float mx = stat;
  if (tid < CDIM) p[tid] = __expf(p[tid] - mx);
  __syncthreads();
  if (tid == 0) { float s = 0.f; for (int i = 0; i < CDIM; ++i) s += p[i]; stat = 1.0f / s; }
  __syncthreads();
  const float inv = stat;
  if (tid < CDIM) p[tid] *= inv;
  __syncthreads();
  if (tid < EDIM) {
    float a = 0.f;
    for (int c = 0; c < CDIM; ++c) a += p[c] * emb1[c * EDIM + tid];
    bf16 v = f2bf(a);
    xdec[(size_t)row * K_XDEC + ROI + tid] = v;
    objE[(size_t)row * K_E + tid] = v;
  } else if (tid < K_E) {
    objE[(size_t)row * K_E + tid] = f2bf(0.0f);
  }
}

// ---------------- pos / box_inp / pair_inp small MLPs ----------------
__global__ __launch_bounds__(128)
void k_small(const float* __restrict__ box, const float* __restrict__ ovl,
             const float* pw1, const float* pb1, const float* pg, const float* pbeta,
             const float* pw2, const float* pb2,
             const float* bw, const float* bb, const float* bg, const float* bbeta,
             const float* ow, const float* ob, const float* og, const float* obeta,
             bf16* __restrict__ xdec, bf16* __restrict__ bi)
{
  __shared__ float bx[9], ov[6], t1[32];
  const int row = blockIdx.x, tid = threadIdx.x;
  if (tid < 9) bx[tid] = box[row * 9 + tid];
  if (tid < 6) ov[tid] = ovl[row * 6 + tid];
  __syncthreads();
  if (tid < 32) {
    float u = pb1[tid];
    for (int k = 0; k < 9; ++k) u += bx[k] * pw1[k * 32 + tid];
    t1[tid] = u * pg[tid] + pbeta[tid];                     // BN (eval), no relu yet
  }
  __syncthreads();
  { // pos = relu(bn(...) @ pw2 + pb2)
    float u = pb2[tid];
    for (int k = 0; k < 32; ++k) u += t1[k] * pw2[k * 128 + tid];
    xdec[(size_t)row * K_XDEC + ROI + EDIM + tid] = f2bf(fmaxf(u, 0.0f));
  }
  { // box_inp = relu(bn(box @ bw + bb))
    float u = bb[tid];
    for (int k = 0; k < 9; ++k) u += bx[k] * bw[k * 128 + tid];
    bi[(size_t)row * 512 + 256 + tid] = f2bf(fmaxf(u * bg[tid] + bbeta[tid], 0.0f));
  }
  { // pair_inp = relu(bn(ovl @ ow + ob))
    float u = ob[tid];
    for (int k = 0; k < 6; ++k) u += ov[k] * ow[k * 128 + tid];
    bi[(size_t)row * 512 + 384 + tid] = f2bf(fmaxf(u * og[tid] + obeta[tid], 0.0f));
  }
  if (tid < 24) xdec[(size_t)row * K_XDEC + 4424 + tid] = f2bf(0.0f);  // obj_pre pad
}

// ---------------- BiLSTM cell (both directions), Hh=256 ----------------
template<bool F32OUT>
__global__ __launch_bounds__(256)
void k_bicell(const float* __restrict__ gxf, const float* __restrict__ gxb, int t,
              float* __restrict__ cf, float* __restrict__ cb,
              bf16* __restrict__ hbf_f, bf16* __restrict__ hbf_b,
              void* __restrict__ dA, int ldA, int offA,
              void* __restrict__ dB, int ldB, int offB)
{
  const int b = blockIdx.x, dir = blockIdx.y, j = threadIdx.x;
  const int tt = dir ? (TDIM - 1 - t) : t;
  const float* g = (dir ? gxb : gxf) + (size_t)(b * TDIM + tt) * NG_HH;
  float* cs = (dir ? cb : cf) + b * HH + j;
  const float gi = g[j], gf = g[HH + j], gg = g[2 * HH + j], go = g[3 * HH + j];
  float c = sigm(gf) * cs[0] + sigm(gi) * tanhf(gg);
  cs[0] = c;
  const float h = sigm(go) * tanhf(c);
  (dir ? hbf_b : hbf_f)[b * HH + j] = f2bf(h);
  const size_t r = (size_t)(b * TDIM + tt);
  const int col = dir * HH + j;
  if (F32OUT) {
    ((float*)dA)[r * ldA + offA + col] = h;
  } else {
    ((bf16*)dA)[r * ldA + offA + col] = f2bf(h);
    if (dB) ((bf16*)dB)[r * ldB + offB + col] = f2bf(h);
  }
}

// ---------------- decoder cell (H=512) ----------------
__global__ __launch_bounds__(512)
void k_dec_cell(const float* __restrict__ gdec, int t,
                float* __restrict__ c, float* __restrict__ hdec, bf16* __restrict__ eh)
{
  const int b = blockIdx.x, j = threadIdx.x;
  const float* g = gdec + (size_t)(b * TDIM + t) * NG_H;
  const float gi = g[j], gf = g[HDIM + j], gg = g[2 * HDIM + j], go = g[3 * HDIM + j];
  float cc = sigm(gf) * c[b * HDIM + j] + sigm(gi) * tanhf(gg);
  c[b * HDIM + j] = cc;
  const float h = sigm(go) * tanhf(cc);
  hdec[b * HDIM + j] = h;
  eh[(size_t)b * K_EH + K_E + j] = f2bf(h);                 // h-part of [e|h]
}

// ---------------- decoder output head: dist, argmax, next embedding ----------------
__global__ __launch_bounds__(256)
void k_dec_out(const float* __restrict__ hdec, const float* __restrict__ outw,
               const float* __restrict__ outb, const float* __restrict__ dec_emb,
               int t, float* __restrict__ dists_out, int* __restrict__ preds,
               bf16* __restrict__ eh)
{
  __shared__ float d[CDIM];
  __shared__ int sp;
  const int b = blockIdx.x, j = threadIdx.x;
  if (j < CDIM) {
    float s = outb[j];
    const float* h = hdec + b * HDIM;
    for (int k = 0; k < HDIM; ++k) s += h[k] * outw[k * CDIM + j];
    d[j] = s;
    dists_out[(size_t)(b * TDIM + t) * CDIM + j] = s;
  }
  __syncthreads();
  if (j == 0) {
    int best = 0; float bv = d[0];
    for (int i = 1; i < CDIM; ++i) if (d[i] > bv) { bv = d[i]; best = i; }
    sp = best;
    preds[b * TDIM + t] = best;
  }
  __syncthreads();
  if (j < K_E) {
    const float e = (j < EDIM) ? dec_emb[(size_t)(sp + 1) * EDIM + j] : 0.0f;
    eh[(size_t)b * K_EH + j] = f2bf(e);
  }
}

__global__ void k_gather_e2(const int* __restrict__ preds, const float* __restrict__ emb2,
                            bf16* __restrict__ ebuf)
{
  const int row = blockIdx.x, j = threadIdx.x;
  if (j >= K_E) return;
  const int p = preds[row];
  const float v = (j < EDIM) ? emb2[(size_t)p * EDIM + j] : 0.0f;
  ebuf[(size_t)row * K_E + j] = f2bf(v);
}

// ---------------- vision prior vc[b,i,j] ----------------
__global__ __launch_bounds__(256)
void k_vc(const float* __restrict__ sub, const float* __restrict__ obj,
          const float* __restrict__ vpw, const float* __restrict__ vpb,
          float* __restrict__ out)
{
  __shared__ float ss[TDIM], so[TDIM];
  const int b = blockIdx.x, tid = threadIdx.x;
  if (tid < TDIM) {
    const float* s = sub + (size_t)(b * TDIM + tid) * HDIM;
    float a = 0.f; for (int h = 0; h < HDIM; ++h) a += s[h] * vpw[h];
    ss[tid] = a;
  } else if (tid < 2 * TDIM) {
    const int j = tid - TDIM;
    const float* s = obj + (size_t)(b * TDIM + j) * HDIM;
    float a = 0.f; for (int h = 0; h < HDIM; ++h) a += s[h] * vpw[HDIM + h];
    so[j] = a;
  }
  __syncthreads();
  const float bb = vpb[0];
  for (int idx = tid; idx < TDIM * TDIM; idx += 256) {
    const int i = idx >> 6, j = idx & 63;
    const float* si = sub + (size_t)(b * TDIM + i) * HDIM;
    const float* oj = obj + (size_t)(b * TDIM + j) * HDIM;
    float m = 0.f;
    for (int h = 0; h < HDIM; ++h) m += si[h] * vpw[2 * HDIM + h] * oj[h];
    out[(size_t)b * TDIM * TDIM + idx] = sigm(ss[i] + so[j] + m + bb);
  }
}

__global__ void k_zero(unsigned* __restrict__ p, long nWords)
{
  long i = (long)blockIdx.x * blockDim.x + threadIdx.x;
  if (i < nWords) p[i] = 0u;
}

__global__ void k_init_e0(const float* __restrict__ dec_emb, bf16* __restrict__ eh)
{
  const int b = blockIdx.x, j = threadIdx.x;
  if (j < K_E) eh[(size_t)b * K_EH + j] = f2bf(j < EDIM ? dec_emb[j] : 0.0f);
}

// =====================================================================================
extern "C" void kernel_launch(void* const* d_in, const int* in_sizes, int n_in,
                              void* d_out, int out_size, void* d_ws, size_t ws_size,
                              hipStream_t stream)
{
  (void)in_sizes; (void)n_in; (void)out_size; (void)ws_size;
  // ---- inputs (setup_inputs order) ----
  const float* x        = (const float*)d_in[0];
  const float* dists_in = (const float*)d_in[1];
  const float* box_info = (const float*)d_in[2];
  const float* ovl_info = (const float*)d_in[3];
  const float* emb1     = (const float*)d_in[4];
  const float* emb2     = (const float*)d_in[5];
  const float* dec_emb  = (const float*)d_in[6];
  const float* pos_w1 = (const float*)d_in[7];  const float* pos_b1 = (const float*)d_in[8];
  const float* pos_g  = (const float*)d_in[9];  const float* pos_be = (const float*)d_in[10];
  const float* pos_w2 = (const float*)d_in[11]; const float* pos_b2 = (const float*)d_in[12];
  const float* box_w  = (const float*)d_in[13]; const float* box_b  = (const float*)d_in[14];
  const float* box_g  = (const float*)d_in[15]; const float* box_be = (const float*)d_in[16];
  const float* ovl_w  = (const float*)d_in[17]; const float* ovl_b  = (const float*)d_in[18];
  const float* ovl_g  = (const float*)d_in[19]; const float* ovl_be = (const float*)d_in[20];
  const float* red_obj_w = (const float*)d_in[21]; const float* red_obj_b = (const float*)d_in[22];
  const float* red_emb_w = (const float*)d_in[23]; const float* red_emb_b = (const float*)d_in[24];
  const float* spre_w = (const float*)d_in[25]; const float* spre_b = (const float*)d_in[26];
  const float* ssub_w = (const float*)d_in[27]; const float* ssub_b = (const float*)d_in[28];
  const float* sobj_w = (const float*)d_in[29]; const float* sobj_b = (const float*)d_in[30];
  const float* vp_w = (const float*)d_in[31];   const float* vp_b = (const float*)d_in[32];
  const float* octx_wf = (const float*)d_in[33]; const float* octx_bf = (const float*)d_in[34];
  const float* octx_wb = (const float*)d_in[35]; const float* octx_bb = (const float*)d_in[36];
  const float* dec_w = (const float*)d_in[37];   const float* dec_b = (const float*)d_in[38];
  const float* out_w = (const float*)d_in[39];   const float* out_b = (const float*)d_in[40];
  const float* ectx_wf = (const float*)d_in[41]; const float* ectx_bf = (const float*)d_in[42];
  const float* ectx_wb = (const float*)d_in[43]; const float* ectx_bb = (const float*)d_in[44];

  // ---- outputs ----
  float* dists_out = (float*)d_out;                          // [B,T,C]
  float* edge_out  = dists_out + (size_t)BT * CDIM;          // [B,T,512]
  float* vc_out    = edge_out  + (size_t)BT * HDIM;          // [B,T,T]

  // ---- workspace carve ----
  size_t off = 0;
  auto carve = [&](size_t bytes) {
    void* p = (char*)d_ws + off;
    off = (off + bytes + 255) & ~(size_t)255;
    return p;
  };
  bf16* xdec  = (bf16*)carve((size_t)BT * K_XDEC * 2);
  bf16* xedge = (bf16*)carve((size_t)BT * K_XEDGE * 2);
  bf16* objE  = (bf16*)carve((size_t)BT * K_E * 2);
  bf16* bi    = (bf16*)carve((size_t)BT * 512 * 2);
  bf16* hpre  = (bf16*)carve((size_t)BT * 512 * 2);
  float* subP = (float*)carve((size_t)BT * 512 * 4);
  float* objP = (float*)carve((size_t)BT * 512 * 4);
  bf16* ebuf  = (bf16*)carve((size_t)BT * K_E * 2);
  float* gxof = (float*)carve((size_t)BT * NG_HH * 4);
  float* gxob = (float*)carve((size_t)BT * NG_HH * 4);
  float* gdec = (float*)carve((size_t)BT * NG_H * 4);
  float* gxef = (float*)carve((size_t)BT * NG_HH * 4);
  float* gxeb = (float*)carve((size_t)BT * NG_HH * 4);
  // bf16-transposed weights [N, Kpad]
  bf16* wRedObj = (bf16*)carve((size_t)128 * 4096 * 2);
  bf16* wRedEmb = (bf16*)carve((size_t)128 * K_E * 2);
  bf16* wSpre   = (bf16*)carve((size_t)512 * 512 * 2);
  bf16* wSsub   = (bf16*)carve((size_t)512 * 512 * 2);
  bf16* wSobj   = (bf16*)carve((size_t)512 * 512 * 2);
  bf16* wOfx    = (bf16*)carve((size_t)NG_HH * K_OBJPRE * 2);
  bf16* wOfh    = (bf16*)carve((size_t)NG_HH * HH * 2);
  bf16* wObx    = (bf16*)carve((size_t)NG_HH * K_OBJPRE * 2);
  bf16* wObh    = (bf16*)carve((size_t)NG_HH * HH * 2);
  bf16* wDecX   = (bf16*)carve((size_t)NG_H * K_XDEC * 2);
  bf16* wDecEH  = (bf16*)carve((size_t)NG_H * K_EH * 2);
  bf16* wEfx    = (bf16*)carve((size_t)NG_HH * K_XEDGE * 2);
  bf16* wEfe    = (bf16*)carve((size_t)NG_HH * K_E * 2);
  bf16* wEfh    = (bf16*)carve((size_t)NG_HH * HH * 2);
  bf16* wEbx    = (bf16*)carve((size_t)NG_HH * K_XEDGE * 2);
  bf16* wEbe    = (bf16*)carve((size_t)NG_HH * K_E * 2);
  bf16* wEbh    = (bf16*)carve((size_t)NG_HH * HH * 2);
  // recurrent state (zeroed every call)
  size_t stateStart = off;
  float* c_of = (float*)carve((size_t)BDIM * HH * 4);
  float* c_ob = (float*)carve((size_t)BDIM * HH * 4);
  float* c_ef = (float*)carve((size_t)BDIM * HH * 4);
  float* c_eb = (float*)carve((size_t)BDIM * HH * 4);
  float* cdec = (float*)carve((size_t)BDIM * HDIM * 4);
  float* hdec = (float*)carve((size_t)BDIM * HDIM * 4);
  bf16* hbf_of = (bf16*)carve((size_t)BDIM * HH * 2);
  bf16* hbf_ob = (bf16*)carve((size_t)BDIM * HH * 2);
  bf16* hbf_ef = (bf16*)carve((size_t)BDIM * HH * 2);
  bf16* hbf_eb = (bf16*)carve((size_t)BDIM * HH * 2);
  bf16* eh_bf  = (bf16*)carve((size_t)BDIM * K_EH * 2);
  int*  preds  = (int*)carve((size_t)BT * 4);
  size_t stateBytes = off - stateStart;

  // ---- launch helpers ----
  auto convtr = [&](const float* src, bf16* dst, int dstLd, int colOff,
                    int Kc, int Ktot, int N, int srcLd, int row0) {
    long total = (long)N * Ktot;
    k_convtr<<<dim3((unsigned)((total + 255) / 256)), 256, 0, stream>>>(
        src, dst, dstLd, colOff, Kc, Ktot, N, srcLd, row0);
  };
  // M % 64 == 0, N % 128 == 0, K % 32 == 0 at every call site.
  auto gemm_bf = [&](const bf16* A, int lda, const bf16* Wt, int ldw, const float* bias,
                     bf16* C, int ldc, int M, int N, int K) {
    dim3 g((unsigned)(N / 128), (unsigned)(M / 64));
    k_gemm<true, false><<<g, 256, 0, stream>>>(A, lda, Wt, ldw, bias, (void*)C, ldc, K);
  };
  auto gemm_f32 = [&](const bf16* A, int lda, const bf16* Wt, int ldw, const float* bias,
                      float* C, int ldc, int M, int N, int K) {
    dim3 g((unsigned)(N / 128), (unsigned)(M / 64));
    k_gemm<false, false><<<g, 256, 0, stream>>>(A, lda, Wt, ldw, bias, (void*)C, ldc, K);
  };
  auto gemm_acc = [&](const bf16* A, int lda, const bf16* Wt, int ldw,
                      float* C, int ldc, int M, int N, int K) {
    dim3 g((unsigned)(N / 128), (unsigned)(M / 64));
    k_gemm<false, true><<<g, 256, 0, stream>>>(A, lda, Wt, ldw, nullptr, (void*)C, ldc, K);
  };

  // ---- phase 0: weight conversion (f32 [K,N] -> bf16 [N,Kpad]) ----
  convtr(red_obj_w, wRedObj, 4096, 0, 4096, 4096, 128, 128, 0);
  convtr(red_emb_w, wRedEmb, K_E, 0, EDIM, K_E, 128, 128, 0);
  convtr(spre_w, wSpre, 512, 0, 512, 512, HDIM, HDIM, 0);
  convtr(ssub_w, wSsub, 512, 0, 512, 512, HDIM, HDIM, 0);
  convtr(sobj_w, wSobj, 512, 0, 512, 512, HDIM, HDIM, 0);
  convtr(octx_wf, wOfx, K_OBJPRE, 0, 4424, K_OBJPRE, NG_HH, NG_HH, 0);
  convtr(octx_wf, wOfh, HH, 0, HH, HH, NG_HH, NG_HH, 4424);
  convtr(octx_wb, wObx, K_OBJPRE, 0, 4424, K_OBJPRE, NG_HH, NG_HH, 0);
  convtr(octx_wb, wObh, HH, 0, HH, HH, NG_HH, NG_HH, 4424);
  convtr(dec_w, wDecX, K_XDEC, 0, 4424, K_OBJPRE, NG_H, NG_H, 0);        // obj_pre part (+pad)
  convtr(dec_w, wDecX, K_XDEC, K_OBJPRE, HDIM, HDIM, NG_H, NG_H, 4424);  // enc part
  convtr(dec_w, wDecEH, K_EH, 0, EDIM, K_E, NG_H, NG_H, 4936);           // e part (+pad)
  convtr(dec_w, wDecEH, K_EH, K_E, HDIM, HDIM, NG_H, NG_H, 5136);        // h part
  convtr(ectx_wf, wEfe, K_E, 0, EDIM, K_E, NG_HH, NG_HH, 0);
  convtr(ectx_wf, wEfx, K_XEDGE, 0, K_XEDGE, K_XEDGE, NG_HH, NG_HH, EDIM);
  convtr(ectx_wf, wEfh, HH, 0, HH, HH, NG_HH, NG_HH, EDIM + K_XEDGE);
  convtr(ectx_wb, wEbe, K_E, 0, EDIM, K_E, NG_HH, NG_HH, 0);
  convtr(ectx_wb, wEbx, K_XEDGE, 0, K_XEDGE, K_XEDGE, NG_HH, NG_HH, EDIM);
  convtr(ectx_wb, wEbh, HH, 0, HH, HH, NG_HH, NG_HH, EDIM + K_XEDGE);

  // ---- phase 1: data prep ----
  k_copy_x<<<dim3((unsigned)(((long)BT * ROI + 255) / 256)), 256, 0, stream>>>(x, xdec, xedge);
  k_softmax_embed<<<BT, 256, 0, stream>>>(dists_in, emb1, xdec, objE);
  k_small<<<BT, 128, 0, stream>>>(box_info, ovl_info,
                                  pos_w1, pos_b1, pos_g, pos_be, pos_w2, pos_b2,
                                  box_w, box_b, box_g, box_be,
                                  ovl_w, ovl_b, ovl_g, ovl_be, xdec, bi);
  {
    long w = (long)(stateBytes / 4);
    k_zero<<<dim3((unsigned)((w + 255) / 256)), 256, 0, stream>>>((unsigned*)((char*)d_ws + stateStart), w);
  }
  k_init_e0<<<BDIM, 256, 0, stream>>>(dec_emb, eh_bf);

  // ---- phase 2: tree-score branch + vision prior ----
  gemm_bf(xedge, K_XEDGE, wRedObj, 4096, red_obj_b, bi, 512, BT, 128, 4096);          // x reduction
  gemm_bf(objE, K_E, wRedEmb, K_E, red_emb_b, bi + 128, 512, BT, 128, K_E);           // emb reduction
  gemm_bf(bi, 512, wSpre, 512, spre_b, hpre, 512, BT, 512, 512);
  gemm_f32(hpre, 512, wSsub, 512, ssub_b, subP, 512, BT, 512, 512);
  gemm_f32(hpre, 512, wSobj, 512, sobj_b, objP, 512, BT, 512, 512);
  k_vc<<<BDIM, 256, 0, stream>>>(subP, objP, vp_w, vp_b, vc_out);

  // ---- phase 3: obj BiLSTM ----
  gemm_f32(xdec, K_XDEC, wOfx, K_OBJPRE, octx_bf, gxof, NG_HH, BT, NG_HH, K_OBJPRE);
  gemm_f32(xdec, K_XDEC, wObx, K_OBJPRE, octx_bb, gxob, NG_HH, BT, NG_HH, K_OBJPRE);
  for (int t = 0; t < TDIM; ++t) {
    gemm_acc(hbf_of, HH, wOfh, HH, gxof + (size_t)t * NG_HH, TDIM * NG_HH, BDIM, NG_HH, HH);
    gemm_acc(hbf_ob, HH, wObh, HH, gxob + (size_t)(TDIM - 1 - t) * NG_HH, TDIM * NG_HH, BDIM, NG_HH, HH);
    k_bicell<false><<<dim3(BDIM, 2), 256, 0, stream>>>(
        gxof, gxob, t, c_of, c_ob, hbf_of, hbf_ob,
        (void*)xdec, K_XDEC, K_OBJPRE, (void*)xedge, K_XEDGE, ROI);
  }

  // ---- phase 4: decoder LSTM with argmax feedback ----
  gemm_f32(xdec, K_XDEC, wDecX, K_XDEC, dec_b, gdec, NG_H, BT, NG_H, K_XDEC);
  for (int t = 0; t < TDIM; ++t) {
    gemm_acc(eh_bf, K_EH, wDecEH, K_EH, gdec + (size_t)t * NG_H, TDIM * NG_H, BDIM, NG_H, K_EH);
    k_dec_cell<<<BDIM, 512, 0, stream>>>(gdec, t, cdec, hdec, eh_bf);
    k_dec_out<<<BDIM, 256, 0, stream>>>(hdec, out_w, out_b, dec_emb, t, dists_out, preds, eh_bf);
  }

  // ---- phase 5: edge BiLSTM ----
  k_gather_e2<<<BT, 256, 0, stream>>>(preds, emb2, ebuf);
  gemm_f32(xedge, K_XEDGE, wEfx, K_XEDGE, ectx_bf, gxef, NG_HH, BT, NG_HH, K_XEDGE);
  gemm_f32(xedge, K_XEDGE, wEbx, K_XEDGE, ectx_bb, gxeb, NG_HH, BT, NG_HH, K_XEDGE);
  gemm_acc(ebuf, K_E, wEfe, K_E, gxef, NG_HH, BT, NG_HH, K_E);
  gemm_acc(ebuf, K_E, wEbe, K_E, gxeb, NG_HH, BT, NG_HH, K_E);
  for (int t = 0; t < TDIM; ++t) {
    gemm_acc(hbf_ef, HH, wEfh, HH, gxef + (size_t)t * NG_HH, TDIM * NG_HH, BDIM, NG_HH, HH);
    gemm_acc(hbf_eb, HH, wEbh, HH, gxeb + (size_t)(TDIM - 1 - t) * NG_HH, TDIM * NG_HH, BDIM, NG_HH, HH);
    k_bicell<true><<<dim3(BDIM, 2), 256, 0, stream>>>(
        gxef, gxeb, t, c_ef, c_eb, hbf_ef, hbf_eb,
        (void*)edge_out, HDIM, 0, nullptr, 0, 0);
  }
}
